// DeformableCrossAttention_14620068675731
// MI455X (gfx1250) — matrix operand order repaired and checked by reading, not compiled, not run on hardware
//
#include <hip/hip_runtime.h>
#include <hip/hip_bf16.h>
#include <math.h>

// ---------------------------------------------------------------------------
// Problem constants (from reference): b=2, n_q=1024, n_kv=2048, d=256,
// H=8, K=4, NN=4, c_=32.
// ---------------------------------------------------------------------------
#define B_SZ     2
#define NQ       1024
#define NKV      2048
#define DMODEL   256
#define HEADS    8
#define KSAMP    4
#define NNBR     4
#define CHEAD    32          // DMODEL / HEADS
#define PCOLS    96          // H*K*2 (=64 off) + H*K (=32 attn logits)
#define MBLK     4           // M-tiles per wave (register blocking)

typedef __attribute__((ext_vector_type(16))) __bf16 v16bf;
typedef __attribute__((ext_vector_type(8)))  float  v8f;

// ---------------------------------------------------------------------------
// Register-blocked WMMA GEMM core: one wave computes MBLK consecutive 16-row
// tiles of a 16-column output slab, sharing the B fragment across all MBLK
// WMMAs per K-step (bf16 inputs, f32 accumulate), wave32.
//
// CDNA5 layouts (ISA 7.12.2):
//   A 16x32 bf16: lane l (m=l&15, h=l>>4): elem e in [0,8)  -> K = h*8 + e
//                                          elem e in [8,16) -> K = 16 + h*8 + (e-8)
//   B 32x16 bf16: lane l (n=l&15, h=l>>4): elem e -> same K as A (h*8+e | 16+h*8+(e-8))
//   C/D 16x16 f32: vgpr r, lane l -> row = h*8 + r, col = l&15
// ---------------------------------------------------------------------------
__device__ __forceinline__ void wmma_tile4_bf16(const float* __restrict__ Ablk,
                                                long lda,
                                                const float* __restrict__ Bcol,
                                                long ldb,
                                                int Ktot, int lane,
                                                v8f acc[MBLK]) {
    const int h = lane >> 4;
    const int m = lane & 15;
    for (int k0 = 0; k0 < Ktot; k0 += 32) {
        // B fragment: loaded ONCE, reused by all MBLK WMMAs below.
        v16bf bm;
#pragma unroll
        for (int e = 0; e < 16; ++e) {
            const int kk = (e < 8) ? (h << 3) + e : 16 + (h << 3) + (e - 8); // K = h*8+e | 16+h*8+(e-8)
            bm[e] = (__bf16)Bcol[(size_t)(k0 + kk) * ldb];
        }
#pragma unroll
        for (int t = 0; t < MBLK; ++t) {
            const float* Ar = Ablk + (size_t)(t * 16 + m) * lda + k0 + h * 8;
            if (k0 + 32 < Ktot) {
                __builtin_prefetch(Ar + 32, 0, 1);     // global_prefetch_b8
            }
            v16bf a;
#pragma unroll
            for (int e = 0; e < 8; ++e)  a[e]     = (__bf16)Ar[e];       // K = h*8+e
#pragma unroll
            for (int e = 0; e < 8; ++e)  a[8 + e] = (__bf16)Ar[16 + e];  // K = 16+h*8+e
            acc[t] = __builtin_amdgcn_wmma_f32_16x16x32_bf16(
                         /*neg_a=*/false, a, /*neg_b=*/false, bm,
                         /*c_mod=*/(short)0, acc[t],
                         /*reuse_a=*/false, /*reuse_b=*/false);
        }
    }
}

// ---------------------------------------------------------------------------
// Kernel 1: P = query @ [W_off | W_attn] + [b_off | b_attn]
// query: [B*NQ, 256] fp32 ; P: [B*NQ, 96] fp32
// wave-tiles: (128/4) Mblks x 6 Ntiles = 192 waves
// ---------------------------------------------------------------------------
__global__ void k_gemm_qproj(const float* __restrict__ query,
                             const float* __restrict__ W_off,
                             const float* __restrict__ b_off,
                             const float* __restrict__ W_attn,
                             const float* __restrict__ b_attn,
                             float* __restrict__ P) {
    const int tilesN = PCOLS / 16;                            // 6
    const int nWaves = (B_SZ * NQ / (16 * MBLK)) * tilesN;    // 32*6 = 192
    const int wave   = blockIdx.x * blockDim.y + threadIdx.y;
    if (wave >= nWaves) return;
    const int tMb  = wave / tilesN;
    const int tN   = wave % tilesN;
    const int lane = threadIdx.x;
    const int h    = lane >> 4;
    const int m    = lane & 15;
    const int col  = tN * 16 + m;

    const float* Bcol;
    long  ldb;
    float bias;
    if (col < 64) { Bcol = W_off  + col;        ldb = 64; bias = b_off[col]; }
    else          { Bcol = W_attn + (col - 64); ldb = 32; bias = b_attn[col - 64]; }

    v8f acc[MBLK] = {};
    wmma_tile4_bf16(query + (size_t)tMb * (16 * MBLK) * DMODEL, DMODEL,
                    Bcol, ldb, DMODEL, lane, acc);
#pragma unroll
    for (int t = 0; t < MBLK; ++t) {
#pragma unroll
        for (int r = 0; r < 8; ++r) {
            const int row = tMb * (16 * MBLK) + t * 16 + h * 8 + r;
            P[(size_t)row * PCOLS + col] = acc[t][r] + bias;
        }
    }
}

// ---------------------------------------------------------------------------
// Kernel 2: values = key_value @ W_v + b_v, stored head-transposed:
// values_t[bh, kv, c] = V[bb, kv, h*32+c]   (bh = bb*H + h)
// wave-tiles: (256/4) Mblks x 16 Ntiles = 1024 waves
// ---------------------------------------------------------------------------
__global__ void k_gemm_values(const float* __restrict__ key_value,
                              const float* __restrict__ W_v,
                              const float* __restrict__ b_v,
                              float* __restrict__ values_t) {
    const int tilesN = DMODEL / 16;                            // 16
    const int nWaves = (B_SZ * NKV / (16 * MBLK)) * tilesN;    // 64*16 = 1024
    const int wave   = blockIdx.x * blockDim.y + threadIdx.y;
    if (wave >= nWaves) return;
    const int tMb  = wave / tilesN;
    const int tN   = wave % tilesN;
    const int lane = threadIdx.x;
    const int h    = lane >> 4;
    const int m    = lane & 15;
    const int col  = tN * 16 + m;

    v8f acc[MBLK] = {};
    wmma_tile4_bf16(key_value + (size_t)tMb * (16 * MBLK) * DMODEL, DMODEL,
                    W_v + col, DMODEL, DMODEL, lane, acc);
    const int hh = col >> 5;        // head
    const int cc = col & 31;        // channel in head
    const float bias = b_v[col];
#pragma unroll
    for (int t = 0; t < MBLK; ++t) {
#pragma unroll
        for (int r = 0; r < 8; ++r) {
            const int row = tMb * (16 * MBLK) + t * 16 + h * 8 + r;  // bb*NKV + kv
            const int bb  = row >> 11;
            const int kv  = row & (NKV - 1);
            values_t[(((size_t)(bb * HEADS + hh)) * NKV + kv) * CHEAD + cc] =
                acc[t][r] + bias;
        }
    }
}

// ---------------------------------------------------------------------------
// Kernel 3: attn = softmax over K of logits (P columns 64..95)
// attn layout: [B*NQ, H, K]
// ---------------------------------------------------------------------------
__global__ void k_softmax_attn(const float* __restrict__ P,
                               float* __restrict__ attn) {
    const int t = blockIdx.x * blockDim.x + threadIdx.x;    // (row, h)
    if (t >= B_SZ * NQ * HEADS) return;
    const int row = t >> 3;
    const int h   = t & 7;
    float l[KSAMP];
    float mx = -3.0e38f;
#pragma unroll
    for (int k = 0; k < KSAMP; ++k) {
        l[k] = P[(size_t)row * PCOLS + 64 + h * KSAMP + k];
        mx = fmaxf(mx, l[k]);
    }
    float s = 0.f;
#pragma unroll
    for (int k = 0; k < KSAMP; ++k) { l[k] = expf(l[k] - mx); s += l[k]; }
    const float inv = 1.0f / s;
#pragma unroll
    for (int k = 0; k < KSAMP; ++k)
        attn[(size_t)t * KSAMP + k] = l[k] * inv;
}

// ---------------------------------------------------------------------------
// Kernel 4: per (bh, q, k): brute-force top-4 nearest kv points, Shepard
// softmax over NN, multiply by attn weight -> combined weights + indices.
// kv_pos for this batch cached in LDS (16 KB of the 320 KB WGP pool).
// grid: (NQ/64, B*H), block: 256 threads = 64 q * 4 k
// ---------------------------------------------------------------------------
__global__ void k_topk(const float* __restrict__ query_pos,
                       const float* __restrict__ kv_pos,
                       const float* __restrict__ P,
                       const float* __restrict__ attn,
                       const float* __restrict__ shepard,
                       int*   __restrict__ idx_ws,
                       float* __restrict__ w_ws) {
    __shared__ float skx[NKV];
    __shared__ float sky[NKV];
    const int bh = blockIdx.y;
    const int b  = bh >> 3;
    const int h  = bh & 7;
    for (int i = threadIdx.x; i < NKV; i += blockDim.x) {
        skx[i] = kv_pos[((size_t)b * NKV + i) * 2 + 0];
        sky[i] = kv_pos[((size_t)b * NKV + i) * 2 + 1];
    }
    __syncthreads();

    const int t = threadIdx.x;
    const int q = blockIdx.x * 64 + (t >> 2);
    const int k = t & 3;
    const size_t row = (size_t)b * NQ + q;

    const float lx = query_pos[row * 2 + 0] + P[row * PCOLS + h * 8 + k * 2 + 0];
    const float ly = query_pos[row * 2 + 1] + P[row * PCOLS + h * 8 + k * 2 + 1];

    float bd0 = 3.0e38f, bd1 = 3.0e38f, bd2 = 3.0e38f, bd3 = 3.0e38f;
    int   bi0 = 0,       bi1 = 0,       bi2 = 0,       bi3 = 0;
    for (int kv = 0; kv < NKV; ++kv) {
        const float dx = lx - skx[kv];
        const float dy = ly - sky[kv];
        const float d  = dx * dx + dy * dy;
        if (d < bd3) {
            if (d < bd2) {
                bd3 = bd2; bi3 = bi2;
                if (d < bd1) {
                    bd2 = bd1; bi2 = bi1;
                    if (d < bd0) { bd1 = bd0; bi1 = bi0; bd0 = d; bi0 = kv; }
                    else         { bd1 = d;   bi1 = kv; }
                } else { bd2 = d; bi2 = kv; }
            } else { bd3 = d; bi3 = kv; }
        }
    }

    const float power = fmaxf(shepard[0], 0.0f) + 1e-6f;
    float dd[NNBR] = { bd0, bd1, bd2, bd3 };
    int   ii[NNBR] = { bi0, bi1, bi2, bi3 };
    float a[NNBR];
    float mx = -3.0e38f;
#pragma unroll
    for (int n = 0; n < NNBR; ++n) {
        const float dist = sqrtf(dd[n]) + 1e-6f;
        a[n] = -power * dist;
        mx = fmaxf(mx, a[n]);
    }
    float s = 0.f;
#pragma unroll
    for (int n = 0; n < NNBR; ++n) { a[n] = expf(a[n] - mx); s += a[n]; }
    const float ak  = attn[(row * HEADS + h) * KSAMP + k];
    const float scl = ak / s;

    const size_t base = (((size_t)bh * NQ + q) * KSAMP + k) * NNBR;
#pragma unroll
    for (int n = 0; n < NNBR; ++n) {
        idx_ws[base + n] = ii[n];
        w_ws[base + n]   = a[n] * scl;
    }
}

// ---------------------------------------------------------------------------
// Kernel 5: gather + combine: mid[b,q, h*32+c] = sum_{k,n} w * values_t[bh,idx,c]
// one thread per (bh, q, c) -> 16*1024*32 = 524288 threads
// ---------------------------------------------------------------------------
__global__ void k_gather(const int*   __restrict__ idx_ws,
                         const float* __restrict__ w_ws,
                         const float* __restrict__ values_t,
                         float* __restrict__ mid) {
    const int t = blockIdx.x * blockDim.x + threadIdx.x;
    if (t >= B_SZ * HEADS * NQ * CHEAD) return;
    const int c  = t & (CHEAD - 1);
    const int q  = (t >> 5) & (NQ - 1);
    const int bh = t >> 15;
    const int b  = bh >> 3;
    const int h  = bh & 7;

    const size_t base  = ((size_t)bh * NQ + q) * (KSAMP * NNBR);
    const size_t vbase = (size_t)bh * NKV * CHEAD;
    float acc = 0.f;
#pragma unroll
    for (int j = 0; j < KSAMP * NNBR; ++j) {
        const int   id = idx_ws[base + j];
        const float w  = w_ws[base + j];
        acc += w * values_t[vbase + (size_t)id * CHEAD + c];
    }
    mid[(((size_t)b * NQ + q) * DMODEL) + h * CHEAD + c] = acc;
}

// ---------------------------------------------------------------------------
// Kernel 6: out = mid @ W_out + b_out   [2048,256]x[256,256]
// wave-tiles: (128/4) Mblks x 16 Ntiles = 512 waves
// ---------------------------------------------------------------------------
__global__ void k_gemm_out(const float* __restrict__ mid,
                           const float* __restrict__ W_out,
                           const float* __restrict__ b_out,
                           float* __restrict__ out) {
    const int tilesN = DMODEL / 16;                           // 16
    const int nWaves = (B_SZ * NQ / (16 * MBLK)) * tilesN;    // 32*16 = 512
    const int wave   = blockIdx.x * blockDim.y + threadIdx.y;
    if (wave >= nWaves) return;
    const int tMb  = wave / tilesN;
    const int tN   = wave % tilesN;
    const int lane = threadIdx.x;
    const int h    = lane >> 4;
    const int m    = lane & 15;
    const int col  = tN * 16 + m;

    v8f acc[MBLK] = {};
    wmma_tile4_bf16(mid + (size_t)tMb * (16 * MBLK) * DMODEL, DMODEL,
                    W_out + col, DMODEL, DMODEL, lane, acc);
    const float bias = b_out[col];
#pragma unroll
    for (int t = 0; t < MBLK; ++t) {
#pragma unroll
        for (int r = 0; r < 8; ++r) {
            const int row = tMb * (16 * MBLK) + t * 16 + h * 8 + r;
            out[(size_t)row * DMODEL + col] = acc[t][r] + bias;
        }
    }
}

// ---------------------------------------------------------------------------
// Host-side launch
// ---------------------------------------------------------------------------
extern "C" void kernel_launch(void* const* d_in, const int* in_sizes, int n_in,
                              void* d_out, int out_size, void* d_ws, size_t ws_size,
                              hipStream_t stream) {
    const float* query     = (const float*)d_in[0];   // [2,1024,256]
    const float* query_pos = (const float*)d_in[1];   // [2,1024,2]
    const float* key_value = (const float*)d_in[2];   // [2,2048,256]
    const float* kv_pos    = (const float*)d_in[3];   // [2,2048,2]
    const float* W_off     = (const float*)d_in[4];   // [256,64]
    const float* b_off     = (const float*)d_in[5];   // [64]
    const float* W_attn    = (const float*)d_in[6];   // [256,32]
    const float* b_attn    = (const float*)d_in[7];   // [32]
    const float* W_v       = (const float*)d_in[8];   // [256,256]
    const float* b_v       = (const float*)d_in[9];   // [256]
    const float* W_out     = (const float*)d_in[10];  // [256,256]
    const float* b_out     = (const float*)d_in[11];  // [256]
    const float* shepard   = (const float*)d_in[12];  // [1]
    float* out = (float*)d_out;                       // [2,1024,256]

    // Workspace carve-up (bytes); ~8.4 MB total, trivially inside L2 (192 MB)
    char* ws = (char*)d_ws;
    size_t off = 0;
    float* P        = (float*)(ws + off); off += (size_t)B_SZ * NQ * PCOLS * sizeof(float);
    float* attn     = (float*)(ws + off); off += (size_t)B_SZ * NQ * HEADS * KSAMP * sizeof(float);
    float* values_t = (float*)(ws + off); off += (size_t)B_SZ * HEADS * NKV * CHEAD * sizeof(float);
    int*   idx_ws   = (int*)  (ws + off); off += (size_t)B_SZ * HEADS * NQ * KSAMP * NNBR * sizeof(int);
    float* w_ws     = (float*)(ws + off); off += (size_t)B_SZ * HEADS * NQ * KSAMP * NNBR * sizeof(float);
    float* mid      = (float*)(ws + off); off += (size_t)B_SZ * NQ * DMODEL * sizeof(float);
    (void)ws_size; (void)in_sizes; (void)n_in; (void)out_size;

    const dim3 wblk(32, 4);   // 4 wave32 waves per block, one 64x16 slab each

    // 1) P = query @ [W_off|W_attn] + bias   (192 waves)
    k_gemm_qproj<<<dim3(192 / 4), wblk, 0, stream>>>(query, W_off, b_off, W_attn, b_attn, P);

    // 2) values_t = head-transposed(key_value @ W_v + b_v)   (1024 waves)
    k_gemm_values<<<dim3(1024 / 4), wblk, 0, stream>>>(key_value, W_v, b_v, values_t);

    // 3) attn softmax over K
    k_softmax_attn<<<dim3((B_SZ * NQ * HEADS + 255) / 256), dim3(256), 0, stream>>>(P, attn);

    // 4) top-4 NN + Shepard weights  (grid: q-chunks x bh)
    k_topk<<<dim3(NQ / 64, B_SZ * HEADS), dim3(256), 0, stream>>>(
        query_pos, kv_pos, P, attn, shepard, idx_ws, w_ws);

    // 5) gather + combine -> mid
    k_gather<<<dim3((B_SZ * HEADS * NQ * CHEAD) / 256), dim3(256), 0, stream>>>(
        idx_ws, w_ws, values_t, mid);

    // 6) out = mid @ W_out + b_out   (512 waves)
    k_gemm_out<<<dim3(512 / 4), wblk, 0, stream>>>(mid, W_out, b_out, out);
}
